// MultibandSpectroTopographicTokenizer_34359738590
// MI455X (gfx1250) — matrix operand order repaired and checked by reading, not compiled, not run on hardware
//
#include <hip/hip_runtime.h>
#include <hip/hip_bf16.h>
#include <math.h>

typedef __attribute__((ext_vector_type(2)))  float  v2f;
typedef __attribute__((ext_vector_type(8)))  float  v8f;
typedef __attribute__((ext_vector_type(8)))  __bf16 v8bf;
typedef __attribute__((ext_vector_type(16))) __bf16 v16bf;
typedef __attribute__((ext_vector_type(4)))  unsigned int v4u;
typedef __attribute__((ext_vector_type(4)))  int v4i;
typedef __attribute__((ext_vector_type(8)))  int v8i;

#define NB    16
#define NCH_  64
#define NT_   15000
#define WS_   500
#define ST_   250
#define NW_   59
#define NF_   251
#define DM_   512
#define EPSF  1e-6f
#define LNEPS 1e-5f
#define NROWS (NB*NW_*NCH_)     /* 60416 */
#define NBW   (NB*NW_)          /* 944   */
#define KST   (WS_/4)           /* 125   */
#define BASN  (KST*16*2*16*2)   /* 128000: [s][nt][half][n16][e2] */

// LDS padding: stride % 64 banks != 0 so 16 row-parallel accesses rotate banks
#define AS_STR 516   /* f32  */
#define HS_STR 520   /* bf16 */
#define CS_STR 516   /* f32  */

#if defined(__gfx1250__) && __has_builtin(__builtin_amdgcn_tensor_load_to_lds) && \
    __has_builtin(__builtin_amdgcn_s_wait_tensorcnt)
#define USE_TDM 1
#else
#define USE_TDM 0
#endif

// ---------------------------------------------------------------------------
// DFT basis, pre-swizzled into V_WMMA_F32_16X16X4_F32 B-fragment order.
// Layout: idx = (((s*16+nt)*2+half)*16+n16)*2+e ; k = 4s+2*half+e ; n = nt*16+n16
// ---------------------------------------------------------------------------
__global__ void basis_kernel(float* __restrict__ cosP, float* __restrict__ sinP) {
  int idx = blockIdx.x * blockDim.x + threadIdx.x;
  if (idx >= BASN) return;
  int e   =  idx        & 1;
  int n16 = (idx >> 1)  & 15;
  int hf  = (idx >> 5)  & 1;
  int nt  = (idx >> 6)  & 15;
  int s   =  idx >> 10;
  int k = 4*s + 2*hf + e;
  int n = nt*16 + n16;
  float cv = 0.f, sv = 0.f;
  if (n < NF_) {
    int kn = (k * n) % WS_;                      // exact angle reduction
    double ang = 6.283185307179586 * (double)kn / (double)WS_;
    cv = (float)cos(ang);
    sv = (float)sin(ang);
  }
  cosP[idx] = cv;
  sinP[idx] = sv;
}

// ---------------------------------------------------------------------------
// bf16 weight pack for V_WMMA_F32_16X16X32_BF16 B, contiguous k per lane half (A matches).
// Layout: idx = (((kt*32+nt)*2+half)*16+n)*16+e ; k = kt*32+half*16+e ; col = nt*16+n
// ---------------------------------------------------------------------------
__global__ void pack_kernel(const float* __restrict__ W, int Krows,
                            __bf16* __restrict__ dst, int total) {
  int idx = blockIdx.x * blockDim.x + threadIdx.x;
  if (idx >= total) return;
  int e  =  idx        & 15;
  int n  = (idx >> 4)  & 15;
  int hf = (idx >> 8)  & 1;
  int nt = (idx >> 9)  & 31;
  int kt =  idx >> 14;
  int k   = kt*32 + hf*16 + e;
  int col = nt*16 + n;
  float v = (k < Krows) ? W[(size_t)k * DM_ + col] : 0.f;
  dst[idx] = (__bf16)v;
}

// ---------------------------------------------------------------------------
// Stage 1: frame extract + demean + mask, DFT via f32 WMMA, fused spectral
// reductions -> rowfeat[R][8] = {bp0..bp3, alpha_peak_freq, alpha_peak_score,0,0}
// One block = 16 rows; 4 waves, each owning 64 freq columns (4 N-tiles).
// ---------------------------------------------------------------------------
__global__ __launch_bounds__(128)
void spectral_kernel(const float* __restrict__ x, const unsigned char* __restrict__ mask,
                     const float* __restrict__ cosP, const float* __restrict__ sinP,
                     float* __restrict__ rowfeat) {
  __shared__ __align__(16) float As[16][AS_STR];
  __shared__ float part[4][16][8];
  int tid = threadIdx.x;

  { // cooperative load + demean + channel mask (8 lanes per row)
    int row = tid >> 3, l8 = tid & 7;
    int gr  = blockIdx.x * 16 + row;
    int b   = gr / (NW_ * NCH_);
    int rem = gr - b * (NW_ * NCH_);
    int w   = rem / NCH_;
    int c   = rem - w * NCH_;
    const float* xp = x + ((size_t)b * NCH_ + c) * NT_ + (size_t)w * ST_;
    float s = 0.f;
    for (int t = l8; t < WS_; t += 8) s += xp[t];
    s += __shfl_xor(s, 1, 32); s += __shfl_xor(s, 2, 32); s += __shfl_xor(s, 4, 32);
    float mean  = s * (1.f / WS_);
    float valid = mask[b * NCH_ + c] ? 1.f : 0.f;
    for (int t = l8; t < WS_; t += 8) As[row][t] = (xp[t] - mean) * valid;
  }
  __syncthreads();

  int wave = tid >> 5, L = tid & 31, Lm = L & 15, Lh = L >> 4;
  v8f accRe[4], accIm[4];
#pragma unroll
  for (int t = 0; t < 4; t++)
#pragma unroll
    for (int v = 0; v < 8; v++) { accRe[t][v] = 0.f; accIm[t][v] = 0.f; }

  for (int s = 0; s < KST; s++) {
    int kb = 4 * s + 2 * Lh;                         // A frag: K = 2*Lh + {0,1}
    v2f a = *(const v2f*)&As[Lm][kb];
#pragma unroll
    for (int t = 0; t < 4; t++) {
      int bi = (((s * 16 + (wave * 4 + t)) * 2 + Lh) * 16 + Lm) * 2;
      v2f bc = *(const v2f*)(cosP + bi);
      v2f bs = *(const v2f*)(sinP + bi);
      accRe[t] = __builtin_amdgcn_wmma_f32_16x16x4_f32(false, a, false, bc,
                                                       (short)0, accRe[t], false, false);
      accIm[t] = __builtin_amdgcn_wmma_f32_16x16x4_f32(false, a, false, bs,
                                                       (short)0, accIm[t], false, false);
      if (s + 1 < KST) {                             // near-scope prefetch of next K-step
        __builtin_prefetch(cosP + bi + 1024, 0, 3);
        __builtin_prefetch(sinP + bi + 1024, 0, 3);
      }
    }
  }

  // Fused spectral epilogue: per-row weighted reductions over frequency.
  // q[i][v]: i=0..3 band power, 4 power sum, 5 alpha sum, 6 alpha*freq sum, 7 alpha max
  float q[8][8];
#pragma unroll
  for (int i = 0; i < 8; i++)
#pragma unroll
    for (int v = 0; v < 8; v++) q[i][v] = 0.f;

#pragma unroll
  for (int t = 0; t < 4; t++) {
    int f = wave * 64 + t * 16 + Lm;                 // freq bin (freq = f*0.5 Hz)
    float w0 = (f >= 2  && f < 8 ) ? (1.f / 6.f ) : 0.f;   // 1-4 Hz
    float w1 = (f >= 8  && f < 16) ? (1.f / 8.f ) : 0.f;   // 4-8 Hz
    float w2 = (f >= 16 && f < 26) ? (1.f / 10.f) : 0.f;   // 8-13 Hz
    float w3 = (f >= 26 && f < 60) ? (1.f / 34.f) : 0.f;   // 13-30 Hz
    float wa = (f >= 14 && f <= 27) ? 1.f : 0.f;           // 7-13.5 Hz
    float wf = wa * 0.5f * (float)f;
    float w1f = (f < NF_) ? 1.f : 0.f;
#pragma unroll
    for (int v = 0; v < 8; v++) {
      float re = accRe[t][v], im = accIm[t][v];
      float p = re * re + im * im + EPSF;
      q[0][v] += w0 * p;  q[1][v] += w1 * p;
      q[2][v] += w2 * p;  q[3][v] += w3 * p;
      q[4][v] += w1f * p; q[5][v] += wa * p;
      q[6][v] += wf * p;  q[7][v] = fmaxf(q[7][v], wa * p);
    }
  }
  // cross-lane reduce over the 16 N-lanes in each half
#pragma unroll
  for (int i = 0; i < 8; i++)
#pragma unroll
    for (int v = 0; v < 8; v++) {
      float val = q[i][v];
      for (int m = 1; m < 16; m <<= 1) {
        float o = __shfl_xor(val, m, 32);
        val = (i == 7) ? fmaxf(val, o) : (val + o);
      }
      q[i][v] = val;
    }
  if (Lm == 0) {
#pragma unroll
    for (int v = 0; v < 8; v++)
#pragma unroll
      for (int i = 0; i < 8; i++) part[wave][v + 8 * Lh][i] = q[i][v];
  }
  __syncthreads();
  if (tid < 16) {
    float acc[8];
#pragma unroll
    for (int i = 0; i < 8; i++) {
      float a0 = part[0][tid][i], a1 = part[1][tid][i];
      float a2 = part[2][tid][i], a3 = part[3][tid][i];
      acc[i] = (i == 7) ? fmaxf(fmaxf(a0, a1), fmaxf(a2, a3)) : (a0 + a1 + a2 + a3);
    }
    int gr = blockIdx.x * 16 + tid;
    float asum  = fmaxf(acc[5], EPSF);
    float apf   = acc[6] / asum;
    float pmean = acc[4] * (1.f / (float)NF_);
    float aps   = acc[7] / fmaxf(pmean, EPSF);
    float* rf = rowfeat + (size_t)gr * 8;
    rf[0] = acc[0]; rf[1] = acc[1]; rf[2] = acc[2]; rf[3] = acc[3];
    rf[4] = apf; rf[5] = aps; rf[6] = 0.f; rf[7] = 0.f;
  }
}

// ---------------------------------------------------------------------------
// Stage 2: per-(b,w) channel aggregation -> chanlog[NBW][4], desc input (bf16)
// ---------------------------------------------------------------------------
__device__ __forceinline__ float bsum64(float v, float* t2) {
  v += __shfl_xor(v, 16, 32); v += __shfl_xor(v, 8, 32);
  v += __shfl_xor(v, 4, 32);  v += __shfl_xor(v, 2, 32); v += __shfl_xor(v, 1, 32);
  __syncthreads();
  if ((threadIdx.x & 31) == 0) t2[threadIdx.x >> 5] = v;
  __syncthreads();
  return t2[0] + t2[1];
}

__global__ __launch_bounds__(64)
void aggregate_kernel(const float* __restrict__ rowfeat, const unsigned char* __restrict__ mask,
                      const float* __restrict__ spos,
                      float* __restrict__ chanlog, __bf16* __restrict__ descin) {
  __shared__ float t2[2];
  int bw = blockIdx.x, b = bw / NW_, c = threadIdx.x;
  const float* rf = rowfeat + ((size_t)bw * NCH_ + c) * 8;
  float bp0 = rf[0], bp1 = rf[1], bp2 = rf[2], bp3 = rf[3];
  float tot = fmaxf(bp0 + bp1 + bp2 + bp3, EPSF);
  float validf = mask[b * NCH_ + c] ? 1.f : 0.f;
  float px = spos[(size_t)(b * NCH_ + c) * 2 + 0];
  float py = spos[(size_t)(b * NCH_ + c) * 2 + 1];
  float lmsk = (px < -0.02f) ? validf : 0.f;
  float rmsk = (px >  0.02f) ? validf : 0.f;
  float amsk = (py >  0.02f) ? validf : 0.f;
  float pmsk = (py < -0.02f) ? validf : 0.f;
  float alpha = bp2;

  float cnt = bsum64(validf, t2);
  float ch0 = bsum64(bp0 * validf, t2), ch1 = bsum64(bp1 * validf, t2);
  float ch2 = bsum64(bp2 * validf, t2), ch3 = bsum64(bp3 * validf, t2);
  float rl0 = bsum64(bp0 / tot * validf, t2), rl1 = bsum64(bp1 / tot * validf, t2);
  float rl2 = bsum64(bp2 / tot * validf, t2), rl3 = bsum64(bp3 / tot * validf, t2);
  float ls = bsum64(alpha * lmsk, t2), lc = bsum64(lmsk, t2);
  float rs = bsum64(alpha * rmsk, t2), rc = bsum64(rmsk, t2);
  float as2 = bsum64(alpha * amsk, t2), ac = bsum64(amsk, t2);
  float ps = bsum64(alpha * pmsk, t2), pc = bsum64(pmsk, t2);

  if (c == 0) {
    float ccnt = fmaxf(cnt, 1.f);
    float bm[4]  = { ch0 / ccnt, ch1 / ccnt, ch2 / ccnt, ch3 / ccnt };
    float brm[4] = { rl0 / ccnt, rl1 / ccnt, rl2 / ccnt, rl3 / ccnt };
    float la = ls / fmaxf(lc, 1.f), ra = rs / fmaxf(rc, 1.f);
    float pa = ps / fmaxf(pc, 1.f), aa = as2 / fmaxf(ac, 1.f);
    float asym = (la - ra) / fmaxf(la + ra + EPSF, EPSF);
    float aap  = (pa - aa) / fmaxf(pa + aa + EPSF, EPSF);
    if (!((ac > 0.f) && (pc > 0.f))) aap = 0.f;
    float dsc[11];
#pragma unroll
    for (int k = 0; k < 4; k++) {
      float lg = logf(fmaxf(bm[k], EPSF));
      dsc[k] = lg; dsc[4 + k] = brm[k];
      chanlog[(size_t)bw * 4 + k] = lg;
    }
    dsc[8] = asym; dsc[9] = aap; dsc[10] = brm[0] + brm[1];
    __bf16* d = descin + (size_t)bw * 32;
#pragma unroll
    for (int j = 0; j < 32; j++) d[j] = (__bf16)((j < 11) ? dsc[j] : 0.f);
  }
}

// ---------------------------------------------------------------------------
// Stage 3: assemble 15 features per row -> bf16 [R][32] (zero padded to K=32)
// ---------------------------------------------------------------------------
__global__ void features_kernel(const float* __restrict__ rowfeat, const float* __restrict__ chanlog,
                                const float* __restrict__ spos, __bf16* __restrict__ featbf) {
  int gr = blockIdx.x * blockDim.x + threadIdx.x;
  if (gr >= NROWS) return;
  int bw = gr >> 6, c = gr & 63, b = bw / NW_;
  const float* rf = rowfeat + (size_t)gr * 8;
  float bp[4] = { rf[0], rf[1], rf[2], rf[3] };
  float tot = fmaxf(bp[0] + bp[1] + bp[2] + bp[3], EPSF);
  float f[16];
#pragma unroll
  for (int k = 0; k < 4; k++) {
    f[k] = logf(fmaxf(bp[k], EPSF));
    f[4 + k] = bp[k] / tot;
    f[8 + k] = f[k] - chanlog[(size_t)bw * 4 + k];
  }
  f[12] = rf[4]; f[13] = rf[5];
  f[14] = spos[(size_t)(b * NCH_ + c) * 2 + 1];
  f[15] = 0.f;
  __bf16* o = featbf + (size_t)gr * 32;
#pragma unroll
  for (int j = 0; j < 16; j++) o[j] = (__bf16)f[j];
#pragma unroll
  for (int j = 16; j < 32; j++) o[j] = (__bf16)0.f;
}

// ---------------------------------------------------------------------------
// TDM: DMA one wave's 8KB weight K-slice (1024 x 8B elements, 1-D tile) to LDS.
// D# per CDNA5 ISA: group0 {count=1, lds_addr,
// global_addr, type=2}; group1 {data_size=8B, tensor_dim0=tile_dim0=1024,
// tensor_dim1=tile_dim1=1, stride=1024}; groups 2/3 zero (<=2D tensor).
// This toolchain exposes the 6-arg builtin: (g0, g1, g2, g3, g4, cpol).
// ---------------------------------------------------------------------------
#if USE_TDM
__device__ __forceinline__ void tdm_load_8k(const __bf16* gsrc, unsigned lds_off) {
  unsigned long long ga = (unsigned long long)gsrc;
  v4u g0;
  g0[0] = 1u;                                           // count=1, user mode
  g0[1] = lds_off;                                      // lds_addr (bytes)
  g0[2] = (unsigned)(ga & 0xFFFFFFFFull);               // global_addr[31:0]
  g0[3] = (unsigned)((ga >> 32) & 0x1FFFFFFull)         // global_addr[56:32]
        | 0x80000000u;                                  // type=2 ("image")
  v8i g1;
  g1[0] = 0x30000;                                      // data_size=3 (8B)
  g1[1] = (int)(1024u << 16);                           // tensor_dim0 = 1024
  g1[2] = (int)(1u << 16);                              // tensor_dim1 = 1
  g1[3] = (int)(1024u << 16);                           // tile_dim0 = 1024
  g1[4] = 1;                                            // tile_dim1 = 1
  g1[5] = 1024;                                         // tensor_dim0_stride
  g1[6] = 0; g1[7] = 0;
  v4i z4 = {0, 0, 0, 0};
  v8i z8 = {0, 0, 0, 0, 0, 0, 0, 0};
  __builtin_amdgcn_tensor_load_to_lds(g0, g1, z4, z4, z8, 0);
}
#endif

// ---------------------------------------------------------------------------
// Fused MLP: D = (gelu(A@W1+b1))@W2+b2, optional LayerNorm, bf16 WMMA.
// One block = 16 rows; 4 waves x 8 N-tiles = 512 cols. Hidden stays in LDS.
// W2 K-slices double-buffered into LDS by the Tensor Data Mover (per wave,
// no cross-wave sharing -> TENSORcnt-only sync, no barriers in the K loop).
// ---------------------------------------------------------------------------
__global__ __launch_bounds__(128)
void mlp_kernel(const __bf16* __restrict__ A, const __bf16* __restrict__ W1p,
                const float* __restrict__ b1, const __bf16* __restrict__ W2p,
                const float* __restrict__ b2, const float* __restrict__ lng,
                const float* __restrict__ lnb, int doLN, float* __restrict__ out) {
  __shared__ __align__(16) __bf16 Hs[16][HS_STR];
  __shared__ __align__(16) float  Cs[16][CS_STR];
#if USE_TDM
  __shared__ __align__(64) __bf16 Wbuf[2][4][8 * 512];  // 64 KB double buffer
#endif
  int tid = threadIdx.x, wave = tid >> 5, L = tid & 31, Lm = L & 15, Lh = L >> 4;
  int row0 = blockIdx.x * 16;
  union AF { v16bf v; v8bf h[2]; };

  v8f zero;
#pragma unroll
  for (int v = 0; v < 8; v++) zero[v] = 0.f;

  // ---- GEMM1 (K=32, one WMMA) + bias + exact GELU -> Hs ----
  AF a1;
  a1.h[0] = *(const v8bf*)(A + (size_t)(row0 + Lm) * 32 + Lh * 16);
  a1.h[1] = *(const v8bf*)(A + (size_t)(row0 + Lm) * 32 + Lh * 16 + 8);
#pragma unroll
  for (int t = 0; t < 8; t++) {
    int nt = wave * 8 + t;
    v16bf bf = *(const v16bf*)(W1p + (size_t)((nt * 2 + Lh) * 16 + Lm) * 16);
    v8f h = __builtin_amdgcn_wmma_f32_16x16x32_bf16(false, a1.v, false, bf,
                                                    (short)0, zero, false, false);
    int col = nt * 16 + Lm;
    float bias = b1[col];
#pragma unroll
    for (int v = 0; v < 8; v++) {
      float z = h[v] + bias;
      float g = 0.5f * z * (1.f + erff(z * 0.70710678118654752f));
      Hs[v + 8 * Lh][col] = (__bf16)g;
    }
  }
  __syncthreads();

  // ---- GEMM2 (K=512, 16 WMMA K-steps per N-tile) ----
  v8f acc[8];
#pragma unroll
  for (int t = 0; t < 8; t++)
#pragma unroll
    for (int v = 0; v < 8; v++) acc[t][v] = 0.f;

#if USE_TDM
  unsigned ldsb[2];
  ldsb[0] = (unsigned)(size_t)(void*)&Wbuf[0][wave][0];
  ldsb[1] = (unsigned)(size_t)(void*)&Wbuf[1][wave][0];
  tdm_load_8k(W2p + (size_t)(8 * wave) * 512, ldsb[0]);        // kt=0 slice
#endif
  for (int kt = 0; kt < 16; kt++) {
#if USE_TDM
    if (kt < 15) {
      tdm_load_8k(W2p + (size_t)((kt + 1) * 32 + 8 * wave) * 512, ldsb[(kt + 1) & 1]);
      __builtin_amdgcn_s_wait_tensorcnt(1);                    // slice kt landed
    } else {
      __builtin_amdgcn_s_wait_tensorcnt(0);
    }
    const __bf16* wb = &Wbuf[kt & 1][wave][0];
#endif
    AF a2;
    a2.h[0] = *(const v8bf*)&Hs[Lm][kt * 32 + Lh * 16];
    a2.h[1] = *(const v8bf*)&Hs[Lm][kt * 32 + Lh * 16 + 8];
#pragma unroll
    for (int t = 0; t < 8; t++) {
#if USE_TDM
      v16bf bf = *(const v16bf*)(wb + (size_t)(((t * 2 + Lh) * 16 + Lm) * 16));
#else
      int nt = wave * 8 + t;
      v16bf bf = *(const v16bf*)(W2p + (size_t)(((kt * 32 + nt) * 2 + Lh) * 16 + Lm) * 16);
#endif
      acc[t] = __builtin_amdgcn_wmma_f32_16x16x32_bf16(false, a2.v, false, bf,
                                                       (short)0, acc[t], false, false);
    }
  }
#pragma unroll
  for (int t = 0; t < 8; t++) {
    int nt = wave * 8 + t, col = nt * 16 + Lm;
    float bias = b2[col];
#pragma unroll
    for (int v = 0; v < 8; v++) Cs[v + 8 * Lh][col] = acc[t][v] + bias;
  }
  __syncthreads();

  if (!doLN) {
    for (int i = tid; i < 16 * DM_; i += 128) {
      int r = i >> 9, col = i & (DM_ - 1);
      out[(size_t)(row0 + r) * DM_ + col] = Cs[r][col];
    }
  } else {
    int r = tid >> 3, l8 = tid & 7;
    float s = 0.f, s2 = 0.f;
    for (int col = l8; col < DM_; col += 8) { float v = Cs[r][col]; s += v; s2 += v * v; }
    s  += __shfl_xor(s, 1, 32);  s  += __shfl_xor(s, 2, 32);  s  += __shfl_xor(s, 4, 32);
    s2 += __shfl_xor(s2, 1, 32); s2 += __shfl_xor(s2, 2, 32); s2 += __shfl_xor(s2, 4, 32);
    float mu = s * (1.f / DM_);
    float var = s2 * (1.f / DM_) - mu * mu;
    float rstd = rsqrtf(var + LNEPS);
    for (int col = l8; col < DM_; col += 8) {
      float v = Cs[r][col];
      out[(size_t)(row0 + r) * DM_ + col] = (v - mu) * rstd * lng[col] + lnb[col];
    }
  }
}

// ---------------------------------------------------------------------------
extern "C" void kernel_launch(void* const* d_in, const int* in_sizes, int n_in,
                              void* d_out, int out_size, void* d_ws, size_t ws_size,
                              hipStream_t stream) {
  const float* x    = (const float*)d_in[0];
  const float* spos = (const float*)d_in[1];
  const unsigned char* mask = (const unsigned char*)d_in[2];
  const float* Wf1 = (const float*)d_in[3];
  const float* bf1 = (const float*)d_in[4];
  const float* Wf2 = (const float*)d_in[5];
  const float* bf2 = (const float*)d_in[6];
  const float* lng = (const float*)d_in[7];
  const float* lnb = (const float*)d_in[8];
  const float* Wd1 = (const float*)d_in[9];
  const float* bd1 = (const float*)d_in[10];
  const float* Wd2 = (const float*)d_in[11];
  const float* bd2 = (const float*)d_in[12];
  float* out = (float*)d_out;

  char* ws = (char*)d_ws;
  size_t off = 0;
  auto alloc = [&](size_t bytes) -> void* {
    void* p = ws + off;
    off += (bytes + 255) & ~(size_t)255;
    return p;
  };
  float*  cosP    = (float*) alloc((size_t)BASN * sizeof(float));
  float*  sinP    = (float*) alloc((size_t)BASN * sizeof(float));
  float*  rowfeat = (float*) alloc((size_t)NROWS * 8 * sizeof(float));
  float*  chanlog = (float*) alloc((size_t)NBW * 4 * sizeof(float));
  __bf16* descin  = (__bf16*)alloc((size_t)NBW * 32 * sizeof(__bf16));
  __bf16* featbf  = (__bf16*)alloc((size_t)NROWS * 32 * sizeof(__bf16));
  __bf16* Wf1p    = (__bf16*)alloc((size_t)1  * 16384 * sizeof(__bf16));
  __bf16* Wf2p    = (__bf16*)alloc((size_t)16 * 16384 * sizeof(__bf16));
  __bf16* Wd1p    = (__bf16*)alloc((size_t)1  * 16384 * sizeof(__bf16));
  __bf16* Wd2p    = (__bf16*)alloc((size_t)16 * 16384 * sizeof(__bf16));

  basis_kernel<<<(BASN + 255) / 256, 256, 0, stream>>>(cosP, sinP);
  pack_kernel<<<(16384  + 255) / 256, 256, 0, stream>>>(Wf1, 15,  Wf1p, 16384);
  pack_kernel<<<(262144 + 255) / 256, 256, 0, stream>>>(Wf2, 512, Wf2p, 262144);
  pack_kernel<<<(16384  + 255) / 256, 256, 0, stream>>>(Wd1, 11,  Wd1p, 16384);
  pack_kernel<<<(262144 + 255) / 256, 256, 0, stream>>>(Wd2, 512, Wd2p, 262144);

  spectral_kernel<<<NROWS / 16, 128, 0, stream>>>(x, mask, cosP, sinP, rowfeat);
  aggregate_kernel<<<NBW, 64, 0, stream>>>(rowfeat, mask, spos, chanlog, descin);
  features_kernel<<<(NROWS + 255) / 256, 256, 0, stream>>>(rowfeat, chanlog, spos, featbf);

  // tokens: [R, 512] with LayerNorm
  mlp_kernel<<<NROWS / 16, 128, 0, stream>>>(featbf, Wf1p, bf1, Wf2p, bf2, lng, lnb, 1, out);
  // desc_out: [NBW, 512], no LayerNorm, appended after tokens
  mlp_kernel<<<NBW / 16, 128, 0, stream>>>(descin, Wd1p, bd1, Wd2p, bd2, lng, lnb, 0,
                                           out + (size_t)NROWS * DM_);
}
